// MoRBlock_59184649339132
// MI455X (gfx1250) — matrix operand order repaired and checked by reading, not compiled, not run on hardware
//
#include <hip/hip_runtime.h>
#include <stdint.h>

#define DEV __device__ __forceinline__

constexpr int DM   = 768;    // d_model
constexpr int SEQ  = 2048;
constexpr int NB   = 2;      // batch
constexpr int NTOK = NB * SEQ;           // 4096
constexpr int NH   = 12;
constexpr int DH   = 64;
constexpr int FF   = 2048;
constexpr int KEEPK = 1228;              // int(2048 * 0.6)
constexpr float EPS = 1e-6f;

typedef __attribute__((ext_vector_type(16))) __bf16 v16bf;
typedef __attribute__((ext_vector_type(8)))  float  v8f;
typedef __attribute__((ext_vector_type(4)))  unsigned int u32x4;
typedef __attribute__((ext_vector_type(8)))  int i32x8;
typedef __attribute__((ext_vector_type(4)))  int i32x4;

#if defined(__has_builtin)
#  if __has_builtin(__builtin_amdgcn_tensor_load_to_lds) && \
      __has_builtin(__builtin_amdgcn_s_wait_tensorcnt)
#    define USE_TDM 1
#  endif
#endif
#ifndef USE_TDM
#  define USE_TDM 0
#endif

union Frag {
    v16bf v;
    uint16_t u[16];
    uint4 q[2];
};

DEV uint16_t f2bf(float f) {
    uint32_t u = __float_as_uint(f);
    uint32_t r = u + 0x7fffu + ((u >> 16) & 1u);   // round-to-nearest-even
    return (uint16_t)(r >> 16);
}
DEV float bf2f(uint16_t b) { return __uint_as_float(((uint32_t)b) << 16); }

DEV v8f vzero() { v8f z = {0.f,0.f,0.f,0.f,0.f,0.f,0.f,0.f}; return z; }

DEV v8f wmma_bf16(v16bf a, v16bf b, v8f c) {
    // D(16x16,f32) = A(16x32,bf16) * B(32x16,bf16) + C
    return __builtin_amdgcn_wmma_f32_16x16x32_bf16(
        false, a, false, b, (short)0, c, false, false);
}

DEV float sigmoidf(float x) { return 1.f / (1.f + __expf(-x)); }

#if USE_TDM
// ---------------------------------------------------------------------------
// Tensor Data Mover: load a 2-D bf16 tile (tile1 rows x tile0 cols, row
// stride = `stride` elements) from global memory into LDS at byte offset
// `lds_off`. Descriptor per CDNA5 ISA.
// ---------------------------------------------------------------------------
DEV void tdm_load_tile_bf16(uint32_t lds_off, const uint16_t* gptr,
                            int dim0, int dim1, int stride,
                            int tile0, int tile1) {
    uint64_t ga = (uint64_t)(uintptr_t)gptr;
    u32x4 g0;
    g0[0] = 1u;                                    // count=1, user descriptor
    g0[1] = lds_off;                               // lds_addr [63:32]
    g0[2] = (uint32_t)ga;                          // global_addr [95:64]
    g0[3] = (uint32_t)((ga >> 32) & 0x1FFFFFFu)    // global_addr [120:96]
          | (2u << 30);                            // type=2 ("image")
    i32x8 g1;
    g1[0] = 1 << 16;                               // data_size=1 (2 bytes)
    g1[1] = (dim0 & 0xFFFF) << 16;                 // tensor_dim0[15:0]
    g1[2] = ((unsigned)dim0 >> 16) | ((dim1 & 0xFFFF) << 16);
    g1[3] = ((unsigned)dim1 >> 16) | (tile0 << 16);// tile_dim0
    g1[4] = tile1 & 0xFFFF;                        // tile_dim1 (tile_dim2=0)
    g1[5] = stride;                                // tensor_dim0_stride[31:0]
    g1[6] = 0;
    g1[7] = 0;
    i32x4 z4 = {0, 0, 0, 0};
#if __clang_major__ >= 23
    i32x8 z8 = {0, 0, 0, 0, 0, 0, 0, 0};
    __builtin_amdgcn_tensor_load_to_lds(g0, g1, z4, z4, z8, 0);
#else
    __builtin_amdgcn_tensor_load_to_lds(g0, g1, z4, z4, 0);
#endif
}
#endif

// ---------------------------------------------------------------------------
// fp32 -> bf16 conversion (weights)
// ---------------------------------------------------------------------------
__global__ void k_f32_to_bf16(const float* __restrict__ src,
                              uint16_t* __restrict__ dst, int n) {
    int i = blockIdx.x * 256 + threadIdx.x;
    if (i < n) dst[i] = f2bf(src[i]);
}

// ---------------------------------------------------------------------------
// Per-token: depth router, RMSNorm(g1) -> bf16, token router score
// ---------------------------------------------------------------------------
__global__ void k_router_norm1(const float* __restrict__ hidden,
                               const float* __restrict__ g1,
                               const float* __restrict__ w_mod,
                               const float* __restrict__ b_mod,
                               const float* __restrict__ w_tok,
                               const float* __restrict__ b_tok,
                               const int*   __restrict__ temperature,
                               uint16_t* __restrict__ normed_bf,
                               float* __restrict__ depth_prob_out,
                               float* __restrict__ active,
                               float* __restrict__ token_score) {
    int t = blockIdx.x;
    const float* x = hidden + (size_t)t * DM;
    __shared__ float red[3 * 256];
    float s1 = 0.f, s2 = 0.f, s3 = 0.f;
    for (int i = threadIdx.x; i < DM; i += 256) {
        float v = x[i];
        s1 += v * v;
        s2 += v * w_mod[i];
        s3 += v * g1[i] * w_tok[i];
    }
    red[threadIdx.x] = s1; red[256 + threadIdx.x] = s2; red[512 + threadIdx.x] = s3;
    __syncthreads();
    for (int off = 128; off > 0; off >>= 1) {
        if (threadIdx.x < off) {
            red[threadIdx.x]       += red[threadIdx.x + off];
            red[256 + threadIdx.x] += red[256 + threadIdx.x + off];
            red[512 + threadIdx.x] += red[512 + threadIdx.x + off];
        }
        __syncthreads();
    }
    float ssq = red[0], dmod = red[256], dtok = red[512];
    float rs = rsqrtf(ssq / (float)DM + EPS);
    if (threadIdx.x == 0) {
        float tinv = 1.f / (float)temperature[0];
        float dp = sigmoidf((dmod + b_mod[0]) * tinv);
        depth_prob_out[t] = dp;
        active[t] = (dp > 0.5f) ? 1.f : 0.f;
        token_score[t] = sigmoidf(rs * dtok + b_tok[0]);
    }
    for (int i = threadIdx.x; i < DM; i += 256)
        normed_bf[(size_t)t * DM + i] = f2bf(x[i] * rs * g1[i]);
}

// ---------------------------------------------------------------------------
// top-k keep mask per batch: element kept iff (#scores strictly greater) < K
// ---------------------------------------------------------------------------
__global__ void k_topk_mask(const float* __restrict__ token_score,
                            const float* __restrict__ active,
                            float* __restrict__ token_mask_out,
                            float* __restrict__ rowscale1) {
    int b = blockIdx.x;
    __shared__ float sc[SEQ];
    for (int i = threadIdx.x; i < SEQ; i += blockDim.x)
        sc[i] = token_score[b * SEQ + i];
    __syncthreads();
    for (int i = threadIdx.x; i < SEQ; i += blockDim.x) {
        float s = sc[i];
        int cnt = 0;
        for (int j = 0; j < SEQ; ++j) cnt += (sc[j] > s) ? 1 : 0;
        float m = (cnt < KEEPK) ? 1.f : 0.f;
        token_mask_out[b * SEQ + i] = m;
        rowscale1[b * SEQ + i] = m * active[b * SEQ + i];
    }
}

// ---------------------------------------------------------------------------
// WMMA GEMM: C[M,N] = A[M,K] * B[K,N], bf16 inputs, f32 accumulate.
// Block = 256 threads = 8 waves, 64x64 output tile, wave = 16x32 sub-tile.
// A tile (64x32) is DMA'd into LDS by the Tensor Data Mover (wave 0 issues
// the descriptor, overlap with B staging, sync via s_wait_tensorcnt).
// B is staged transposed in LDS so fragment reads are contiguous 16B ds loads
// matching the wave32 WMMA VGPR layouts.
// ---------------------------------------------------------------------------
enum GemmEpi { EPI_BF16 = 0, EPI_RESID = 1, EPI_GATE = 2, EPI_SWIGLU = 3 };

template <int EPI>
__launch_bounds__(256)
__global__ void k_gemm(const uint16_t* __restrict__ A,
                       const uint16_t* __restrict__ B0,
                       const uint16_t* __restrict__ B1,
                       uint16_t* __restrict__ Cb,
                       float* __restrict__ Cf,
                       const float* __restrict__ resid,
                       const float* __restrict__ rowscale,
                       const float* __restrict__ bias,
                       const uint16_t* __restrict__ Nsrc,
                       int M, int N, int K) {
    __shared__ __align__(16) uint16_t la[64][32];
    __shared__ __align__(16) uint16_t lb[64][32];                       // lb[n][k]
    __shared__ __align__(16) uint16_t lb2[(EPI == EPI_SWIGLU) ? 64 : 1][32];

    const int tid  = threadIdx.x;
    const int lane = tid & 31;
    const int wid  = tid >> 5;
    const int wm   = wid >> 1;          // 0..3  (16-row band)
    const int wn   = wid & 1;           // 0..1  (32-col band)
    const int bm   = blockIdx.y * 64;
    const int bn   = blockIdx.x * 64;
    const int up   = lane >> 4;         // 0/1
    const int lr   = lane & 15;

    // cooperative-load coordinates
    const int br = tid >> 3, bc = (tid & 7) << 3;   // B: 32 rows x 64 cols
#if !USE_TDM
    const int ar = tid >> 2, ac = (tid & 3) << 3;   // A: 64 rows x 32 cols
#endif

    v8f acc0 = vzero(), acc1 = vzero(), acc2 = vzero(), acc3 = vzero();

    for (int kk = 0; kk < K; kk += 32) {
#if USE_TDM
        if (wid == 0) {
            // DMA the 64x32 A tile straight into LDS while everyone else
            // stages B; completion tracked by TENSORcnt.
            tdm_load_tile_bf16((uint32_t)(uintptr_t)&la[0][0],
                               A + (size_t)bm * K + kk,
                               /*dim0=*/K, /*dim1=*/64, /*stride=*/K,
                               /*tile0=*/32, /*tile1=*/64);
        }
#else
        uint4 av = *reinterpret_cast<const uint4*>(A + (size_t)(bm + ar) * K + kk + ac);
        *reinterpret_cast<uint4*>(&la[ar][ac]) = av;
#endif
        if (kk + 32 < K)   // prefetch next B tile -> global_prefetch_b8
            __builtin_prefetch(B0 + (size_t)(kk + 32 + br) * N + bn + bc, 0, 0);

        uint4 bv = *reinterpret_cast<const uint4*>(B0 + (size_t)(kk + br) * N + bn + bc);
        const uint16_t* bp = reinterpret_cast<const uint16_t*>(&bv);
        #pragma unroll
        for (int j = 0; j < 8; ++j) lb[bc + j][br] = bp[j];
        if constexpr (EPI == EPI_SWIGLU) {
            uint4 bw = *reinterpret_cast<const uint4*>(B1 + (size_t)(kk + br) * N + bn + bc);
            const uint16_t* bq = reinterpret_cast<const uint16_t*>(&bw);
            #pragma unroll
            for (int j = 0; j < 8; ++j) lb2[bc + j][br] = bq[j];
        }
#if USE_TDM
        __builtin_amdgcn_s_wait_tensorcnt(0);
#endif
        __syncthreads();

        // A fragment: lane<16 holds row, K 0..15; lane>=16: K 16..31
        Frag af;
        const int arow = wm * 16 + lr;
        const int koff = up ? 16 : 0;
        af.q[0] = *reinterpret_cast<const uint4*>(&la[arow][koff]);
        af.q[1] = *reinterpret_cast<const uint4*>(&la[arow][koff + 8]);

        // B fragment: lane holds one column; lanes<16: K 0..15, lanes>=16: K 16..31
        const int kb = up ? 16 : 0;
        const int col0 = wn * 32 + lr, col1 = col0 + 16;
        Frag bf0, bf1;
        bf0.q[0] = *reinterpret_cast<const uint4*>(&lb[col0][kb]);
        bf0.q[1] = *reinterpret_cast<const uint4*>(&lb[col0][kb + 8]);
        bf1.q[0] = *reinterpret_cast<const uint4*>(&lb[col1][kb]);
        bf1.q[1] = *reinterpret_cast<const uint4*>(&lb[col1][kb + 8]);

        acc0 = wmma_bf16(af.v, bf0.v, acc0);
        acc1 = wmma_bf16(af.v, bf1.v, acc1);
        if constexpr (EPI == EPI_SWIGLU) {
            Frag cf0, cf1;
            cf0.q[0] = *reinterpret_cast<const uint4*>(&lb2[col0][kb]);
            cf0.q[1] = *reinterpret_cast<const uint4*>(&lb2[col0][kb + 8]);
            cf1.q[0] = *reinterpret_cast<const uint4*>(&lb2[col1][kb]);
            cf1.q[1] = *reinterpret_cast<const uint4*>(&lb2[col1][kb + 8]);
            acc2 = wmma_bf16(af.v, cf0.v, acc2);
            acc3 = wmma_bf16(af.v, cf1.v, acc3);
        }
        __syncthreads();
    }

    // epilogue: C/D layout — lane holds column lr; rows r(+8 for upper half)
    const int row0 = bm + wm * 16 + (up ? 8 : 0);
    const int c0 = bn + wn * 32 + lr, c1 = c0 + 16;
    #pragma unroll
    for (int r = 0; r < 8; ++r) {
        const int gm = row0 + r;
        const size_t i0 = (size_t)gm * N + c0;
        const size_t i1 = (size_t)gm * N + c1;
        const float v0 = acc0[r], v1 = acc1[r];
        if constexpr (EPI == EPI_BF16) {
            Cb[i0] = f2bf(v0); Cb[i1] = f2bf(v1);
        } else if constexpr (EPI == EPI_RESID) {
            const float sc = rowscale[gm];
            Cf[i0] = resid[i0] + v0 * sc;
            Cf[i1] = resid[i1] + v1 * sc;
        } else if constexpr (EPI == EPI_GATE) {
            Cb[i0] = f2bf(bf2f(Nsrc[i0]) * sigmoidf(v0 + bias[c0]));
            Cb[i1] = f2bf(bf2f(Nsrc[i1]) * sigmoidf(v1 + bias[c1]));
        } else if constexpr (EPI == EPI_SWIGLU) {
            Cb[i0] = f2bf((v0 * sigmoidf(v0)) * acc2[r]);   // silu(h1)*h3
            Cb[i1] = f2bf((v1 * sigmoidf(v1)) * acc3[r]);
        }
    }
}

// ---------------------------------------------------------------------------
// V transpose: Vt[b][h][d][s] = V[(b*S+s)][h*DH+d]  (so PV B-frags are contiguous)
// ---------------------------------------------------------------------------
__global__ void k_transpose_v(const uint16_t* __restrict__ Vb,
                              uint16_t* __restrict__ Vt) {
    int idx = blockIdx.x * 256 + threadIdx.x;
    const int total = NB * NH * DH * SEQ;
    if (idx >= total) return;
    int s = idx % SEQ;
    int d = (idx / SEQ) % DH;
    int h = (idx / (SEQ * DH)) % NH;
    int b = idx / (SEQ * DH * NH);
    Vt[idx] = Vb[((size_t)(b * SEQ + s)) * DM + h * DH + d];
}

// ---------------------------------------------------------------------------
// Flash attention: one wave per (batch, head, 16-query tile).
// Q*K^T via WMMA (2 per 16-key tile), online softmax, P*V via WMMA (4 per
// 32-key block). Causal mask applied analytically.
// ---------------------------------------------------------------------------
__launch_bounds__(32)
__global__ void k_attn(const uint16_t* __restrict__ Qb,
                       const uint16_t* __restrict__ Kb,
                       const uint16_t* __restrict__ Vt,
                       uint16_t* __restrict__ Ctx) {
    const int qt = blockIdx.x;      // query tile
    const int h  = blockIdx.y;
    const int b  = blockIdx.z;
    const int lane = threadIdx.x;
    const int up = lane >> 4, lr = lane & 15;
    const int tok0 = b * SEQ + qt * 16;

    __shared__ __align__(16) uint16_t pl[16][32];

    // Q A-fragments (dh 0..31 and 32..63)
    const uint16_t* qrow = Qb + (size_t)(tok0 + lr) * DM + h * DH;
    const int koff = up ? 16 : 0;
    Frag qa0, qa1;
    qa0.q[0] = *reinterpret_cast<const uint4*>(qrow + koff);
    qa0.q[1] = *reinterpret_cast<const uint4*>(qrow + koff + 8);
    qa1.q[0] = *reinterpret_cast<const uint4*>(qrow + 32 + koff);
    qa1.q[1] = *reinterpret_cast<const uint4*>(qrow + 32 + koff + 8);

    v8f o[4] = {vzero(), vzero(), vzero(), vzero()};
    float mrow[8], lrow[8];
    #pragma unroll
    for (int r = 0; r < 8; ++r) { mrow[r] = -3e38f; lrow[r] = 0.f; }

    const int nkeys = qt * 16 + 16;                 // causal horizon
    const int kbv = up ? 16 : 0;                    // B-frag K offset per half-wave

    for (int kb = 0; kb < nkeys; kb += 32) {
        // ---- logits for 2 key tiles of 16 ----
        v8f s0 = vzero(), s1 = vzero();
        {
            const uint16_t* kr0 = Kb + (size_t)(b * SEQ + kb + lr) * DM + h * DH;
            const uint16_t* kr1 = Kb + (size_t)(b * SEQ + kb + 16 + lr) * DM + h * DH;
            Frag f;
            f.q[0] = *reinterpret_cast<const uint4*>(kr0 + kbv);
            f.q[1] = *reinterpret_cast<const uint4*>(kr0 + kbv + 8);
            s0 = wmma_bf16(qa0.v, f.v, s0);
            f.q[0] = *reinterpret_cast<const uint4*>(kr0 + 32 + kbv);
            f.q[1] = *reinterpret_cast<const uint4*>(kr0 + 32 + kbv + 8);
            s0 = wmma_bf16(qa1.v, f.v, s0);
            f.q[0] = *reinterpret_cast<const uint4*>(kr1 + kbv);
            f.q[1] = *reinterpret_cast<const uint4*>(kr1 + kbv + 8);
            s1 = wmma_bf16(qa0.v, f.v, s1);
            f.q[0] = *reinterpret_cast<const uint4*>(kr1 + 32 + kbv);
            f.q[1] = *reinterpret_cast<const uint4*>(kr1 + 32 + kbv + 8);
            s1 = wmma_bf16(qa1.v, f.v, s1);
        }
        // ---- online softmax ----
        const int key0 = kb + lr, key1 = kb + 16 + lr;
        float p0[8], p1[8];
        #pragma unroll
        for (int r = 0; r < 8; ++r) {
            const int gq = qt * 16 + (up ? 8 : 0) + r;
            float a0 = s0[r] * 0.125f; if (key0 > gq) a0 = -3e38f;
            float a1 = s1[r] * 0.125f; if (key1 > gq) a1 = -3e38f;
            float mx = fmaxf(a0, a1);
            #pragma unroll
            for (int m = 8; m > 0; m >>= 1) mx = fmaxf(mx, __shfl_xor(mx, m, 32));
            const float mn = fmaxf(mrow[r], mx);
            const float alpha = __expf(mrow[r] - mn);
            const float e0 = __expf(a0 - mn), e1 = __expf(a1 - mn);
            float ps = e0 + e1;
            #pragma unroll
            for (int m = 8; m > 0; m >>= 1) ps += __shfl_xor(ps, m, 32);
            lrow[r] = lrow[r] * alpha + ps;
            mrow[r] = mn;
            p0[r] = e0; p1[r] = e1;
            o[0][r] *= alpha; o[1][r] *= alpha; o[2][r] *= alpha; o[3][r] *= alpha;
        }
        // ---- stage P (16x32) into LDS, re-load in A layout ----
        #pragma unroll
        for (int r = 0; r < 8; ++r) {
            const int m = (up ? 8 : 0) + r;
            pl[m][lr]      = f2bf(p0[r]);
            pl[m][16 + lr] = f2bf(p1[r]);
        }
        __syncthreads();
        Frag pa;
        pa.q[0] = *reinterpret_cast<const uint4*>(&pl[lr][koff]);
        pa.q[1] = *reinterpret_cast<const uint4*>(&pl[lr][koff + 8]);
        // ---- P*V : 4 dim sub-tiles, K=32 keys ----
        #pragma unroll
        for (int dt = 0; dt < 4; ++dt) {
            const int d0 = dt * 16 + lr;
            const uint16_t* vr = Vt + ((size_t)((b * NH + h) * DH + d0)) * SEQ + kb + kbv;
            Frag vf;
            vf.q[0] = *reinterpret_cast<const uint4*>(vr);
            vf.q[1] = *reinterpret_cast<const uint4*>(vr + 8);
            o[dt] = wmma_bf16(pa.v, vf.v, o[dt]);
        }
        __syncthreads();
    }

    #pragma unroll
    for (int dt = 0; dt < 4; ++dt)
        #pragma unroll
        for (int r = 0; r < 8; ++r) {
            const int m = (up ? 8 : 0) + r;
            Ctx[(size_t)(tok0 + m) * DM + h * DH + dt * 16 + lr] =
                f2bf(o[dt][r] / lrow[r]);
        }
}

// ---------------------------------------------------------------------------
// RMSNorm(g2) of mid-hidden -> bf16
// ---------------------------------------------------------------------------
__global__ void k_rmsnorm2(const float* __restrict__ x,
                           const float* __restrict__ g2,
                           uint16_t* __restrict__ out) {
    int t = blockIdx.x;
    const float* row = x + (size_t)t * DM;
    __shared__ float red[256];
    float s = 0.f;
    for (int i = threadIdx.x; i < DM; i += 256) { float v = row[i]; s += v * v; }
    red[threadIdx.x] = s; __syncthreads();
    for (int off = 128; off > 0; off >>= 1) {
        if (threadIdx.x < off) red[threadIdx.x] += red[threadIdx.x + off];
        __syncthreads();
    }
    float rs = rsqrtf(red[0] / (float)DM + EPS);
    for (int i = threadIdx.x; i < DM; i += 256)
        out[(size_t)t * DM + i] = f2bf(row[i] * rs * g2[i]);
}

// ---------------------------------------------------------------------------
extern "C" void kernel_launch(void* const* d_in, const int* in_sizes, int n_in,
                              void* d_out, int out_size, void* d_ws, size_t ws_size,
                              hipStream_t stream) {
    (void)in_sizes; (void)n_in; (void)out_size; (void)ws_size;

    const float* hidden = (const float*)d_in[0];
    const int*   temp   = (const int*)d_in[2];
    const float* wq = (const float*)d_in[3];
    const float* wk = (const float*)d_in[4];
    const float* wv = (const float*)d_in[5];
    const float* wo = (const float*)d_in[6];
    const float* w1 = (const float*)d_in[7];
    const float* w3 = (const float*)d_in[8];
    const float* w2 = (const float*)d_in[9];
    const float* g1 = (const float*)d_in[10];
    const float* g2 = (const float*)d_in[11];
    const float* w_mod = (const float*)d_in[12];
    const float* b_mod = (const float*)d_in[13];
    const float* w_tok = (const float*)d_in[14];
    const float* b_tok = (const float*)d_in[15];
    const float* w_gate = (const float*)d_in[16];
    const float* b_gate = (const float*)d_in[17];

    float* out_hidden = (float*)d_out;
    float* out_dprob  = out_hidden + (size_t)NTOK * DM;
    float* out_mask   = out_dprob + NTOK;

    char* p = (char*)d_ws;
    auto alloc = [&](size_t bytes) -> char* {
        char* r = p; p += (bytes + 255) & ~(size_t)255; return r;
    };
    uint16_t* wq_bf = (uint16_t*)alloc((size_t)DM * DM * 2);
    uint16_t* wk_bf = (uint16_t*)alloc((size_t)DM * DM * 2);
    uint16_t* wv_bf = (uint16_t*)alloc((size_t)DM * DM * 2);
    uint16_t* wo_bf = (uint16_t*)alloc((size_t)DM * DM * 2);
    uint16_t* wg_bf = (uint16_t*)alloc((size_t)DM * DM * 2);
    uint16_t* w1_bf = (uint16_t*)alloc((size_t)DM * FF * 2);
    uint16_t* w3_bf = (uint16_t*)alloc((size_t)DM * FF * 2);
    uint16_t* w2_bf = (uint16_t*)alloc((size_t)FF * DM * 2);
    uint16_t* nrm_bf = (uint16_t*)alloc((size_t)NTOK * DM * 2);
    uint16_t* q_bf  = (uint16_t*)alloc((size_t)NTOK * DM * 2);
    uint16_t* kk_bf = (uint16_t*)alloc((size_t)NTOK * DM * 2);
    uint16_t* v_bf  = (uint16_t*)alloc((size_t)NTOK * DM * 2);
    uint16_t* vt_bf = (uint16_t*)alloc((size_t)NB * NH * DH * SEQ * 2);
    uint16_t* ctx_bf = (uint16_t*)alloc((size_t)NTOK * DM * 2);
    float*    hid_mid = (float*)alloc((size_t)NTOK * DM * 4);
    uint16_t* n2_bf  = (uint16_t*)alloc((size_t)NTOK * DM * 2);
    uint16_t* gated_bf = (uint16_t*)alloc((size_t)NTOK * DM * 2);
    uint16_t* hf_bf  = (uint16_t*)alloc((size_t)NTOK * FF * 2);
    float* active    = (float*)alloc(NTOK * 4);
    float* score     = (float*)alloc(NTOK * 4);
    float* rowscale1 = (float*)alloc(NTOK * 4);

    auto cvt = [&](const float* s, uint16_t* d, int n) {
        k_f32_to_bf16<<<(n + 255) / 256, 256, 0, stream>>>(s, d, n);
    };
    cvt(wq, wq_bf, DM * DM);  cvt(wk, wk_bf, DM * DM);
    cvt(wv, wv_bf, DM * DM);  cvt(wo, wo_bf, DM * DM);
    cvt(w_gate, wg_bf, DM * DM);
    cvt(w1, w1_bf, DM * FF);  cvt(w3, w3_bf, DM * FF);
    cvt(w2, w2_bf, FF * DM);

    // routers + norm1
    k_router_norm1<<<NTOK, 256, 0, stream>>>(hidden, g1, w_mod, b_mod, w_tok, b_tok,
                                             temp, nrm_bf, out_dprob, active, score);
    // top-k keep mask
    k_topk_mask<<<NB, 256, 0, stream>>>(score, active, out_mask, rowscale1);

    // Q/K/V projections
    dim3 gQKV(DM / 64, NTOK / 64);
    k_gemm<EPI_BF16><<<gQKV, 256, 0, stream>>>(nrm_bf, wq_bf, nullptr, q_bf, nullptr,
        nullptr, nullptr, nullptr, nullptr, NTOK, DM, DM);
    k_gemm<EPI_BF16><<<gQKV, 256, 0, stream>>>(nrm_bf, wk_bf, nullptr, kk_bf, nullptr,
        nullptr, nullptr, nullptr, nullptr, NTOK, DM, DM);
    k_gemm<EPI_BF16><<<gQKV, 256, 0, stream>>>(nrm_bf, wv_bf, nullptr, v_bf, nullptr,
        nullptr, nullptr, nullptr, nullptr, NTOK, DM, DM);

    k_transpose_v<<<(NB * NH * DH * SEQ + 255) / 256, 256, 0, stream>>>(v_bf, vt_bf);

    // flash attention
    dim3 gAttn(SEQ / 16, NH, NB);
    k_attn<<<gAttn, 32, 0, stream>>>(q_bf, kk_bf, vt_bf, ctx_bf);

    // output projection + masked residual: hid_mid = hidden + (ctx@wo)*mask*active
    k_gemm<EPI_RESID><<<gQKV, 256, 0, stream>>>(ctx_bf, wo_bf, nullptr, nullptr, hid_mid,
        hidden, rowscale1, nullptr, nullptr, NTOK, DM, DM);

    // norm2
    k_rmsnorm2<<<NTOK, 256, 0, stream>>>(hid_mid, g2, n2_bf);

    // ACM gate: gated = n2 * sigmoid(n2@w_gate + b_gate)
    k_gemm<EPI_GATE><<<gQKV, 256, 0, stream>>>(n2_bf, wg_bf, nullptr, gated_bf, nullptr,
        nullptr, nullptr, b_gate, n2_bf, NTOK, DM, DM);

    // SwiGLU up: hf = silu(gated@w1) * (gated@w3)
    dim3 gUp(FF / 64, NTOK / 64);
    k_gemm<EPI_SWIGLU><<<gUp, 256, 0, stream>>>(gated_bf, w1_bf, w3_bf, hf_bf, nullptr,
        nullptr, nullptr, nullptr, nullptr, NTOK, FF, DM);

    // SwiGLU down + residual: out = hid_mid + (hf@w2)*active
    k_gemm<EPI_RESID><<<gQKV, 256, 0, stream>>>(hf_bf, w2_bf, nullptr, nullptr, out_hidden,
        hid_mid, active, nullptr, nullptr, NTOK, DM, FF);
}
